// SelfAttentionWithLearnableToken_41386304864307
// MI455X (gfx1250) — compile-verified, matrix operand order checked by reading, not run on hardware
//
#include <hip/hip_runtime.h>

// ---------------------------------------------------------------------------
// Problem constants (from reference): B=4, S=4096, D=1024.
// Keys = S+1 (learnable global token prepended), padded to SP (mult of 128).
// ---------------------------------------------------------------------------
constexpr int Dd  = 1024;
constexpr int Ss  = 4096;
constexpr int Bb  = 4;
constexpr int BS  = Bb * Ss;        // 16384 flattened rows
constexpr int NK  = Ss + 1;         // 4097 real keys
constexpr int SP  = 4224;           // padded keys (33 * 128)

typedef __attribute__((ext_vector_type(16))) __bf16 v16bf;
typedef __attribute__((ext_vector_type(8)))  __bf16 v8bf;
typedef __attribute__((ext_vector_type(8)))  float  v8f;

union Frag { v16bf f; v8bf h[2]; };

// ---------------------------------------------------------------------------
// f32 -> bf16 conversion, 8 elements / thread, 16B vector stores
// ---------------------------------------------------------------------------
__global__ __launch_bounds__(256) void cvt_f32_bf16(const float* __restrict__ src,
                                                    __bf16* __restrict__ dst,
                                                    long long n8) {
  long long i = (long long)blockIdx.x * 256 + threadIdx.x;
  if (i >= n8) return;
  const float4* s = (const float4*)(src + i * 8);
  float4 a = s[0], c = s[1];
  v8bf o;
  o[0] = (__bf16)a.x; o[1] = (__bf16)a.y; o[2] = (__bf16)a.z; o[3] = (__bf16)a.w;
  o[4] = (__bf16)c.x; o[5] = (__bf16)c.y; o[6] = (__bf16)c.z; o[7] = (__bf16)c.w;
  *(v8bf*)(dst + i * 8) = o;
}

// ---------------------------------------------------------------------------
// Init: global token into K row 0 / V^T col 0; zero key-padding regions.
// Launch: B*D threads.
// ---------------------------------------------------------------------------
__global__ __launch_bounds__(256) void init_kv(const float* __restrict__ gk,
                                               const float* __restrict__ gv,
                                               __bf16* __restrict__ Kc,
                                               __bf16* __restrict__ Vt) {
  int t = blockIdx.x * blockDim.x + threadIdx.x;   // 0 .. B*D-1
  int b = t >> 10, d = t & (Dd - 1);
  __bf16* kb = Kc + (size_t)b * SP * Dd;           // [SP][D] per batch
  __bf16* vb = Vt + ((size_t)b * Dd + d) * SP;     // [D][SP] per batch
  kb[d] = (__bf16)gk[d];                           // K row 0 = gk
  vb[0] = (__bf16)gv[d];                           // V^T col 0 = gv
  for (int r = NK; r < SP; ++r) kb[(size_t)r * Dd + d] = (__bf16)0.f;
  for (int k = NK; k < SP; ++k) vb[k] = (__bf16)0.f;
}

// ---------------------------------------------------------------------------
// Generic NT GEMM: C[m,n] = sum_k A[m,k] * B[n,k]   (both bf16, K-contiguous)
// Workgroup: 256 thr = 8 waves, 128(M) x 128(N) block, wave tile 32x64.
// Fragments per CDNA5 16-bit A-matrix layout: lane<16 -> K {0..7,16..23},
// lane>=16 -> K {8..15,24..31}; two b128 loads per fragment.
// Epilogue modes: 0 = f32 row-major, 1 = bf16 row-major, 2 = bf16 transposed.
// value = (acc + bias[n]) * scale
// ---------------------------------------------------------------------------
__global__ __launch_bounds__(256) void gemm_bf16_nt(
    const __bf16* __restrict__ A, unsigned long long aBatch, int lda,
    const __bf16* __restrict__ Bm, unsigned long long bBatch, int ldb,
    void* __restrict__ Cp, unsigned long long cBatch, int ldc,
    int Kd, const float* __restrict__ bias, float scale, int mode) {
  const int lane = threadIdx.x & 31;
  const int wave = threadIdx.x >> 5;
  const int hi   = lane >> 4;
  const int lr   = lane & 15;
  const int wm   = wave & 3;      // 4 waves along M
  const int wn   = wave >> 2;     // 2 waves along N
  const int m0   = blockIdx.y * 128 + wm * 32;
  const int n0   = blockIdx.x * 128 + wn * 64;

  const __bf16* Ab = A  + (size_t)blockIdx.z * aBatch;
  const __bf16* Bz = Bm + (size_t)blockIdx.z * bBatch;

  const __bf16* arow[2];
  const __bf16* brow[4];
#pragma unroll
  for (int i = 0; i < 2; ++i) arow[i] = Ab + (size_t)(m0 + 16 * i + lr) * lda;
#pragma unroll
  for (int j = 0; j < 4; ++j) brow[j] = Bz + (size_t)(n0 + 16 * j + lr) * ldb;

  v8f acc[2][4];
#pragma unroll
  for (int i = 0; i < 2; ++i)
#pragma unroll
    for (int j = 0; j < 4; ++j)
      acc[i][j] = (v8f){0.f, 0.f, 0.f, 0.f, 0.f, 0.f, 0.f, 0.f};

  const int koff0 = 8 * hi;        // elements 0..7  : K = k0 + 8*hi + e
  const int koff1 = 16 + 8 * hi;   // elements 8..15 : K = k0 + 16 + 8*hi + e

  for (int k0 = 0; k0 < Kd; k0 += 32) {
    Frag a[2], b[4];
#pragma unroll
    for (int i = 0; i < 2; ++i) {
      a[i].h[0] = *(const v8bf*)(arow[i] + k0 + koff0);
      a[i].h[1] = *(const v8bf*)(arow[i] + k0 + koff1);
    }
#pragma unroll
    for (int j = 0; j < 4; ++j) {
      b[j].h[0] = *(const v8bf*)(brow[j] + k0 + koff0);
      b[j].h[1] = *(const v8bf*)(brow[j] + k0 + koff1);
    }
#pragma unroll
    for (int i = 0; i < 2; ++i)
#pragma unroll
      for (int j = 0; j < 4; ++j)
        acc[i][j] = __builtin_amdgcn_wmma_f32_16x16x32_bf16(
            false, a[i].f, false, b[j].f, (short)0, acc[i][j], false, false);
  }

  // Epilogue. C/D layout: element e of v8f -> row = m0+16i+8*hi+e, col = lr.
  const size_t cb = (size_t)blockIdx.z * cBatch;
#pragma unroll
  for (int i = 0; i < 2; ++i)
#pragma unroll
    for (int j = 0; j < 4; ++j) {
      const int col = n0 + 16 * j + lr;
      const float bv = bias ? bias[col] : 0.f;
#pragma unroll
      for (int e = 0; e < 8; ++e) {
        const int row = m0 + 16 * i + 8 * hi + e;
        const float v = (acc[i][j][e] + bv) * scale;
        if (mode == 0) {
          ((float*)Cp)[cb + (size_t)row * ldc + col] = v;
        } else if (mode == 1) {
          ((__bf16*)Cp)[cb + (size_t)row * ldc + col] = (__bf16)v;
        } else { // transposed store (V^T)
          ((__bf16*)Cp)[cb + (size_t)col * ldc + row] = (__bf16)v;
        }
      }
    }
}

// ---------------------------------------------------------------------------
// Row softmax over NK real keys; writes bf16 weights, zeroes padded keys.
// One 256-thread block per (b,s) row.
// ---------------------------------------------------------------------------
__global__ __launch_bounds__(256) void softmax_rows(const float* __restrict__ Sc,
                                                    __bf16* __restrict__ P) {
  __shared__ float red[256];
  const int tid = threadIdx.x;
  const size_t base = (size_t)blockIdx.x * SP;

  float m = -3.4e38f;
  for (int k = tid; k < NK; k += 256) m = fmaxf(m, Sc[base + k]);
  red[tid] = m;
  __syncthreads();
  for (int o = 128; o > 0; o >>= 1) {
    if (tid < o) red[tid] = fmaxf(red[tid], red[tid + o]);
    __syncthreads();
  }
  const float rowmax = red[0];
  __syncthreads();

  float s = 0.f;
  for (int k = tid; k < NK; k += 256) s += __expf(Sc[base + k] - rowmax);
  red[tid] = s;
  __syncthreads();
  for (int o = 128; o > 0; o >>= 1) {
    if (tid < o) red[tid] += red[tid + o];
    __syncthreads();
  }
  const float inv = 1.f / red[0];

  for (int k = tid; k < SP; k += 256)
    P[base + k] = (k < NK) ? (__bf16)(__expf(Sc[base + k] - rowmax) * inv)
                           : (__bf16)0.f;
}

// ---------------------------------------------------------------------------
// Host-side orchestration
// ---------------------------------------------------------------------------
extern "C" void kernel_launch(void* const* d_in, const int* in_sizes, int n_in,
                              void* d_out, int out_size, void* d_ws, size_t ws_size,
                              hipStream_t stream) {
  const float* x  = (const float*)d_in[0];
  const float* Wq = (const float*)d_in[1];
  const float* bq = (const float*)d_in[2];
  const float* Wk = (const float*)d_in[3];
  const float* bk = (const float*)d_in[4];
  const float* Wv = (const float*)d_in[5];
  const float* bv = (const float*)d_in[6];
  const float* Wo = (const float*)d_in[7];
  const float* bo = (const float*)d_in[8];
  const float* gk = (const float*)d_in[9];
  const float* gv = (const float*)d_in[10];

  // Workspace carving (256B aligned)
  size_t off = 0;
  auto carve = [&](size_t bytes) -> void* {
    void* p = (void*)((char*)d_ws + off);
    off += (bytes + 255) & ~(size_t)255;
    return p;
  };
  __bf16* xb   = (__bf16*)carve((size_t)BS * Dd * 2);        // x in bf16
  __bf16* Wqb  = (__bf16*)carve((size_t)Dd * Dd * 2);
  __bf16* Wkb  = (__bf16*)carve((size_t)Dd * Dd * 2);
  __bf16* Wvb  = (__bf16*)carve((size_t)Dd * Dd * 2);
  __bf16* Wob  = (__bf16*)carve((size_t)Dd * Dd * 2);
  __bf16* Qb   = (__bf16*)carve((size_t)BS * Dd * 2);        // Q * (1/sqrt(D))
  __bf16* Kc   = (__bf16*)carve((size_t)Bb * SP * Dd * 2);   // [B][SP][D]
  __bf16* Vt   = (__bf16*)carve((size_t)Bb * Dd * SP * 2);   // [B][D][SP]
  float*  Scf  = (float*) carve((size_t)Bb * Ss * SP * 4);   // scores f32
  __bf16* Pb   = (__bf16*)carve((size_t)Bb * Ss * SP * 2);   // softmax weights
  __bf16* Atb  = (__bf16*)carve((size_t)BS * Dd * 2);        // attn output bf16
  (void)ws_size; (void)in_sizes; (void)n_in; (void)out_size;

  const float qscale = 0.03125f;  // 1/sqrt(1024)

  // 1) Convert inputs to bf16
  {
    long long n8 = (long long)BS * Dd / 8;
    cvt_f32_bf16<<<(unsigned)((n8 + 255) / 256), 256, 0, stream>>>(x, xb, n8);
    long long w8 = (long long)Dd * Dd / 8;
    unsigned wg = (unsigned)((w8 + 255) / 256);
    cvt_f32_bf16<<<wg, 256, 0, stream>>>(Wq, Wqb, w8);
    cvt_f32_bf16<<<wg, 256, 0, stream>>>(Wk, Wkb, w8);
    cvt_f32_bf16<<<wg, 256, 0, stream>>>(Wv, Wvb, w8);
    cvt_f32_bf16<<<wg, 256, 0, stream>>>(Wo, Wob, w8);
  }

  // 2) Global tokens + padding init
  init_kv<<<(Bb * Dd) / 256, 256, 0, stream>>>(gk, gv, Kc, Vt);

  // 3) Projections (x @ W^T + b), grid.z = batch
  {
    dim3 g(Dd / 128, Ss / 128, Bb);
    // Q (scaled by 1/sqrt(D)), bf16 row-major
    gemm_bf16_nt<<<g, 256, 0, stream>>>(
        xb, (unsigned long long)Ss * Dd, Dd,
        Wqb, 0ull, Dd,
        (void*)Qb, (unsigned long long)Ss * Dd, Dd,
        Dd, bq, qscale, 1);
    // K -> rows 1..S of Kc (row 0 = gk)
    gemm_bf16_nt<<<g, 256, 0, stream>>>(
        xb, (unsigned long long)Ss * Dd, Dd,
        Wkb, 0ull, Dd,
        (void*)(Kc + Dd), (unsigned long long)SP * Dd, Dd,
        Dd, bk, 1.f, 1);
    // V -> transposed store into cols 1..S of Vt (col 0 = gv)
    gemm_bf16_nt<<<g, 256, 0, stream>>>(
        xb, (unsigned long long)Ss * Dd, Dd,
        Wvb, 0ull, Dd,
        (void*)(Vt + 1), (unsigned long long)Dd * SP, SP,
        Dd, (const float*)nullptr, 1.f, 2);
  }

  // 4) scores = Qs @ K^T  ([S x D] x [SP x D]^T per batch), f32
  {
    dim3 g(SP / 128, Ss / 128, Bb);
    gemm_bf16_nt<<<g, 256, 0, stream>>>(
        Qb, (unsigned long long)Ss * Dd, Dd,
        Kc, (unsigned long long)SP * Dd, Dd,
        (void*)Scf, (unsigned long long)Ss * SP, SP,
        Dd, (const float*)nullptr, 1.f, 0);
  }

  // 5) softmax rows -> P (bf16, pads zeroed)
  softmax_rows<<<BS, 256, 0, stream>>>(Scf, Pb);

  // 6) attn = P @ V  ( [S x SP] x [D x SP]^T per batch ), bf16
  {
    dim3 g(Dd / 128, Ss / 128, Bb);
    gemm_bf16_nt<<<g, 256, 0, stream>>>(
        Pb, (unsigned long long)Ss * SP, SP,
        Vt, (unsigned long long)Dd * SP, SP,
        (void*)Atb, (unsigned long long)Ss * Dd, Dd,
        SP, (const float*)nullptr, 1.f, 1);
  }

  // 7) out = attn @ Wo^T + bo, f32
  {
    dim3 g(Dd / 128, Ss / 128, Bb);
    gemm_bf16_nt<<<g, 256, 0, stream>>>(
        Atb, (unsigned long long)Ss * Dd, Dd,
        Wob, 0ull, Dd,
        d_out, (unsigned long long)Ss * Dd, Dd,
        Dd, bo, 1.f, 0);
  }
}
